// GraphSAGE_10763188044561
// MI455X (gfx1250) — matrix operand order repaired and checked by reading, not compiled, not run on hardware
//
#include <hip/hip_runtime.h>

typedef __bf16 v16bf __attribute__((ext_vector_type(16)));
typedef float  v8f   __attribute__((ext_vector_type(8)));

#define NNODES 16384
#define DFEAT  256

// ---------------------------------------------------------------------------
// Prep kernel 1: x [16384][256] f32  ->  xT [256][16384] bf16 (tiled transpose)
// ---------------------------------------------------------------------------
__global__ void sage_prep_xT(const float* __restrict__ x, __bf16* __restrict__ xT) {
    __shared__ __bf16 tile[32][33];
    const int bx = blockIdx.x;   // feature tile  (DFEAT/32 = 8)
    const int by = blockIdx.y;   // node tile     (NNODES/32 = 512)
    const int lx = threadIdx.x;  // 0..31
    const int ly = threadIdx.y;  // 0..7
#pragma unroll
    for (int i = 0; i < 32; i += 8) {
        const int r = by * 32 + ly + i;
        const int c = bx * 32 + lx;
        tile[ly + i][lx] = (__bf16)x[(size_t)r * DFEAT + c];
    }
    __syncthreads();
#pragma unroll
    for (int i = 0; i < 32; i += 8) {
        const int r = by * 32 + lx;          // node index
        const int c = bx * 32 + ly + i;      // feature index
        xT[(size_t)c * NNODES + r] = tile[lx][ly + i];
    }
}

// ---------------------------------------------------------------------------
// Prep kernel 2: W [256][256] f32 -> bf16 (row-major kept: row o = W[o][.])
// ---------------------------------------------------------------------------
__global__ void sage_prep_W(const float* __restrict__ W, __bf16* __restrict__ Wb) {
    const int idx = blockIdx.x * 256 + threadIdx.x;
    Wb[idx] = (__bf16)W[idx];
}

// 0/1 ints -> two packed bf16 (1.0 = 0x3F80): pure integer ops, bit-exact.
__device__ __forceinline__ unsigned pack01(int lo, int hi) {
    return (unsigned)lo * 0x3F80u + (unsigned)hi * 0x3F800000u;
}

// Workgroup barrier that only waits for LDS ops (not in-flight global loads
// headed to VGPRs). Safe for the double-buffered staging pattern below.
__device__ __forceinline__ void lds_barrier() {
    asm volatile("s_wait_dscnt 0x0\n\t"
                 "s_barrier_signal -1\n\t"
                 "s_barrier_wait -1" ::: "memory");
}

// ---------------------------------------------------------------------------
// Main fused kernel: agg = (adj@x)/deg ; out = agg@W^T + b
// grid = NNODES/64 blocks, 256 threads (8 waves of 32).
// Wave w owns output column tiles {2w, 2w+1} x row tiles 0..3.
// ---------------------------------------------------------------------------
__global__ __launch_bounds__(256)
void sage_main(const int*    __restrict__ adj,
               const __bf16* __restrict__ xT,
               const __bf16* __restrict__ Wb,
               const float*  __restrict__ bias,
               float*        __restrict__ out) {
    // A-fragment staging, double buffered, K=64 per step:
    // [buf][kchunk][rowtile][lane][elem]
    __shared__ __align__(32) __bf16 a_stage[2][2][4][32][16];  // 16 KB
    // Normalized agg in A-fragment layout for GEMM-2: [kblock][rowtile][lane][elem]
    __shared__ __align__(32) __bf16 agg_lds[8][4][32][16];     // 32 KB
    __shared__ float red[256];
    __shared__ float inv_deg[64];

    const int tid    = threadIdx.x;
    const int wave   = tid >> 5;        // 0..7
    const int lane   = tid & 31;
    const int khalf  = lane >> 4;       // 0/1 : which half of lanes
    const int lan16  = lane & 15;
    const int rowbase = blockIdx.x * 64;

    // ---- staging role: thread t loads 16 consecutive adj ints of one row ----
    const int s_row   = tid >> 2;                    // 0..63
    const int s_q     = tid & 3;                     // 16-wide chunk of K64
    const int s_tile  = s_row >> 4;                  // row tile 0..3
    const int s_m     = s_row & 15;
    const int s_chunk = s_q >> 1;                    // which 32-K chunk (0/1)
    const int s_eb    = (s_q & 1) << 3;              // 0/8; s_eb<<1 = frag lane-half offset 0/16
    const int* arp = adj + (size_t)(rowbase + s_row) * NNODES + s_q * 16;

    // ---- B fragments (GEMM-1): lane = col n, elems = 16 contiguous K from xT ----
    const __bf16* bp0 = xT + (size_t)((wave * 2 + 0) * 16 + lan16) * NNODES + khalf * 16;
    const __bf16* bp1 = xT + (size_t)((wave * 2 + 1) * 16 + lan16) * NNODES + khalf * 16;

    v8f acc[2][4];
#pragma unroll
    for (int c = 0; c < 2; ++c)
#pragma unroll
        for (int r = 0; r < 4; ++r)
#pragma unroll
            for (int j = 0; j < 8; ++j) acc[c][r][j] = 0.0f;

    int deg_i = 0;

    // ================= GEMM-1 main loop: K = 16384 in steps of 64 =================
    for (int k0 = 0, it = 0; k0 < NNODES; k0 += 64, ++it) {
        const int buf = it & 1;

        // speculative prefetch of the adj stream ~4 steps (1 KB/row) ahead
        __builtin_prefetch((const void*)(arp + k0 + 64 * 4), 0, 0);

        // cooperative load of adj block [64 rows x 64 k] as 4x b128 per thread
        const int4 r0 = *(const int4*)(arp + k0);
        const int4 r1 = *(const int4*)(arp + k0 + 4);
        const int4 r2 = *(const int4*)(arp + k0 + 8);
        const int4 r3 = *(const int4*)(arp + k0 + 12);
        deg_i += r0.x + r0.y + r0.z + r0.w + r1.x + r1.y + r1.z + r1.w
               + r2.x + r2.y + r2.z + r2.w + r3.x + r3.y + r3.z + r3.w;

        // Fragment mapping within this thread's 32-K chunk: local k' = (s_q&1)*16 + j.
        // j=0..7  -> frag lane = m + (s_q&1)*16, elems 0..7
        // j=8..15 -> frag lane = m + (s_q&1)*16, elems 8..15
        uint4 pkA, pkB;
        pkA.x = pack01(r0.x, r0.y);  pkA.y = pack01(r0.z, r0.w);
        pkA.z = pack01(r1.x, r1.y);  pkA.w = pack01(r1.z, r1.w);
        pkB.x = pack01(r2.x, r2.y);  pkB.y = pack01(r2.z, r2.w);
        pkB.z = pack01(r3.x, r3.y);  pkB.w = pack01(r3.z, r3.w);
        *(uint4*)&a_stage[buf][s_chunk][s_tile][s_m + (s_eb << 1)][0] = pkA;
        *(uint4*)&a_stage[buf][s_chunk][s_tile][s_m + (s_eb << 1)][8] = pkB;

        // B fragments straight from L2-resident xT (32B contiguous per lane)
        const v16bf bf00 = *(const v16bf*)(bp0 + k0);
        const v16bf bf10 = *(const v16bf*)(bp1 + k0);
        const v16bf bf01 = *(const v16bf*)(bp0 + k0 + 32);
        const v16bf bf11 = *(const v16bf*)(bp1 + k0 + 32);

        lds_barrier();   // dscnt-only wait: staging visible (double-buffered)

#pragma unroll
        for (int r = 0; r < 4; ++r) {
            const v16bf af0 = *(const v16bf*)&a_stage[buf][0][r][lane][0];
            acc[0][r] = __builtin_amdgcn_wmma_f32_16x16x32_bf16(
                false, af0, false, bf00, (short)0, acc[0][r], false, false);
            acc[1][r] = __builtin_amdgcn_wmma_f32_16x16x32_bf16(
                false, af0, false, bf10, (short)0, acc[1][r], false, false);
        }
#pragma unroll
        for (int r = 0; r < 4; ++r) {
            const v16bf af1 = *(const v16bf*)&a_stage[buf][1][r][lane][0];
            acc[0][r] = __builtin_amdgcn_wmma_f32_16x16x32_bf16(
                false, af1, false, bf01, (short)0, acc[0][r], false, false);
            acc[1][r] = __builtin_amdgcn_wmma_f32_16x16x32_bf16(
                false, af1, false, bf11, (short)0, acc[1][r], false, false);
        }
    }

    // ================= degree reduction: 4 partials per row =================
    red[tid] = (float)deg_i;
    __syncthreads();
    if (tid < 64) {
        const float d = red[4 * tid] + red[4 * tid + 1] + red[4 * tid + 2] + red[4 * tid + 3];
        inv_deg[tid] = 1.0f / d;
    }
    __syncthreads();

    // ===== normalize and scatter agg into A-fragment layout (K-block = wave) =====
    // C/D layout: VGPR j, lane L: m = j + (L>=16)*8, n = L&15
#pragma unroll
    for (int c = 0; c < 2; ++c) {
        const int fl = c * 16 + lan16;                 // feature within this 32-K block
        const int dl = fl & 15;                        // fragment element
        const int ll = (fl >> 4) << 4;                 // fragment lane-half offset
#pragma unroll
        for (int r = 0; r < 4; ++r)
#pragma unroll
            for (int j = 0; j < 8; ++j) {
                const int m = j + (khalf << 3);
                const float v = acc[c][r][j] * inv_deg[r * 16 + m];
                agg_lds[wave][r][m + ll][dl] = (__bf16)v;
            }
    }
    __syncthreads();

    // ================= GEMM-2: out = agg @ W^T + b  (K = 256) =================
    v8f acc2[2][4];
#pragma unroll
    for (int c = 0; c < 2; ++c) {
        const float bv = bias[(wave * 2 + c) * 16 + lan16];   // C[m][n] = b[n]
#pragma unroll
        for (int r = 0; r < 4; ++r)
#pragma unroll
            for (int j = 0; j < 8; ++j) acc2[c][r][j] = bv;
    }

    // B = W^T: frag lane = out-col o, elems = 16 contiguous features = row of W
    const __bf16* wp0 = Wb + (size_t)((wave * 2 + 0) * 16 + lan16) * DFEAT + khalf * 16;
    const __bf16* wp1 = Wb + (size_t)((wave * 2 + 1) * 16 + lan16) * DFEAT + khalf * 16;

#pragma unroll
    for (int kb = 0; kb < 8; ++kb) {
        const v16bf wf0 = *(const v16bf*)(wp0 + kb * 32);
        const v16bf wf1 = *(const v16bf*)(wp1 + kb * 32);
#pragma unroll
        for (int r = 0; r < 4; ++r) {
            const v16bf af = *(const v16bf*)&agg_lds[kb][r][lane][0];
            acc2[0][r] = __builtin_amdgcn_wmma_f32_16x16x32_bf16(
                false, af, false, wf0, (short)0, acc2[0][r], false, false);
            acc2[1][r] = __builtin_amdgcn_wmma_f32_16x16x32_bf16(
                false, af, false, wf1, (short)0, acc2[1][r], false, false);
        }
    }

    // ================= store fp32 output =================
#pragma unroll
    for (int c = 0; c < 2; ++c)
#pragma unroll
        for (int r = 0; r < 4; ++r)
#pragma unroll
            for (int j = 0; j < 8; ++j) {
                const int m = j + (khalf << 3);
                out[(size_t)(rowbase + r * 16 + m) * DFEAT + (wave * 2 + c) * 16 + lan16]
                    = acc2[c][r][j];
            }
}

// ---------------------------------------------------------------------------
extern "C" void kernel_launch(void* const* d_in, const int* in_sizes, int n_in,
                              void* d_out, int out_size, void* d_ws, size_t ws_size,
                              hipStream_t stream) {
    const float* x    = (const float*)d_in[0];
    const int*   adj  = (const int*)d_in[1];
    const float* W    = (const float*)d_in[2];
    const float* bias = (const float*)d_in[3];
    float*       out  = (float*)d_out;

    // workspace: xT bf16 (8 MB) + W bf16 (128 KB)
    __bf16* xT = (__bf16*)d_ws;
    __bf16* Wb = (__bf16*)((char*)d_ws + (size_t)DFEAT * NNODES * sizeof(__bf16));

    sage_prep_xT<<<dim3(DFEAT / 32, NNODES / 32), dim3(32, 8), 0, stream>>>(x, xT);
    sage_prep_W<<<(DFEAT * DFEAT) / 256, 256, 0, stream>>>(W, Wb);
    sage_main<<<NNODES / 64, 256, 0, stream>>>(adj, xT, Wb, bias, out);
}
